// GatedLinearAttention_89507118448798
// MI455X (gfx1250) — matrix operand order repaired and checked by reading, not compiled, not run on hardware
//
#include <hip/hip_runtime.h>
#include <hip/hip_bf16.h>

typedef __attribute__((ext_vector_type(16))) _Float16 v16h;
typedef __attribute__((ext_vector_type(8)))  _Float16 v8h;
typedef __attribute__((ext_vector_type(4)))  _Float16 v4h;
typedef __attribute__((ext_vector_type(8)))  float    v8f;
typedef __attribute__((ext_vector_type(4)))  unsigned int v4u;
typedef __attribute__((ext_vector_type(8)))  int v8i;
typedef __attribute__((ext_vector_type(4)))  int v4i;

#if __has_builtin(__builtin_amdgcn_tensor_load_to_lds) && __has_builtin(__builtin_amdgcn_s_wait_tensorcnt)
#define HAS_TDM 1
#else
#define HAS_TDM 0
#endif

__device__ __forceinline__ v8f wmma_f16(v16h a, v16h b, v8f c) {
  return __builtin_amdgcn_wmma_f32_16x16x32_f16(false, a, false, b, (short)0, c, false, false);
}

// A fragment (16x32, row-major source): two contiguous 16-byte runs per lane.
__device__ __forceinline__ v16h ld_afrag(const _Float16* base, int stride, int row, int k0, int lane) {
  const _Float16* p = base + (row + (lane & 15)) * stride + k0 + (lane >> 4) * 8;
  v8h lo = *(const v8h*)(p);
  v8h hi = *(const v8h*)(p + 16);
  return __builtin_shufflevector(lo, hi, 0, 1, 2, 3, 4, 5, 6, 7, 8, 9, 10, 11, 12, 13, 14, 15);
}
// B fragment (32x16) from K-contiguous-per-column source: two contiguous 16-byte runs per lane.
__device__ __forceinline__ v16h ld_bfrag(const _Float16* base, int stride, int col, int k0, int lane) {
  const _Float16* p = base + (col + (lane & 15)) * stride + k0 + (lane >> 4) * 8;
  v8h lo = *(const v8h*)(p);
  v8h hi = *(const v8h*)(p + 16);
  return __builtin_shufflevector(lo, hi, 0, 1, 2, 3, 4, 5, 6, 7, 8, 9, 10, 11, 12, 13, 14, 15);
}

#if HAS_TDM
// Build D# per CDNA5 ISA §8.3/8.4 and issue a 2-D f16 tile load (DMA) into LDS.
// tile_d0 elems contiguous (dim-0), tile_d1 rows, row stride = stride0 elems.
__device__ __forceinline__ void tdm_load_2d_f16(unsigned lds_off, const _Float16* gaddr,
                                                unsigned tile_d0, unsigned tile_d1,
                                                unsigned tensor_d0, unsigned tensor_d1,
                                                unsigned long stride0) {
  unsigned long ga = (unsigned long)gaddr;
  v4u g0;
  g0.x = 0x1u;                                              // count=1, user mode
  g0.y = lds_off;                                           // lds_addr [63:32]
  g0.z = (unsigned)(ga & 0xFFFFFFFFu);                      // global_addr [95:64]
  g0.w = (unsigned)((ga >> 32) & 0x1FFFFFFu) | (2u << 30);  // addr[56:32] | type=2
  v8i g1;
  g1[0] = (int)(1u << 16);                                  // data_size=1 (2 bytes)
  g1[1] = (int)((tensor_d0 & 0xFFFFu) << 16);               // tensor_dim0[15:0] @ 63:48
  g1[2] = (int)(((tensor_d0 >> 16) & 0xFFFFu) | ((tensor_d1 & 0xFFFFu) << 16));
  g1[3] = (int)(((tensor_d1 >> 16) & 0xFFFFu) | ((tile_d0 & 0xFFFFu) << 16));
  g1[4] = (int)(tile_d1 & 0xFFFFu);                         // tile_dim1 (tile_dim2=0)
  g1[5] = (int)(stride0 & 0xFFFFFFFFu);                     // tensor_dim0_stride[31:0]
  g1[6] = (int)((stride0 >> 32) & 0xFFFFu);                 // stride[47:32]
  g1[7] = 0;
  v4i z4 = {0, 0, 0, 0};
#if defined(__clang_major__) && __clang_major__ >= 23
  v8i z8 = {0, 0, 0, 0, 0, 0, 0, 0};
  __builtin_amdgcn_tensor_load_to_lds(g0, g1, z4, z4, z8, 0);
#else
  __builtin_amdgcn_tensor_load_to_lds(g0, g1, z4, z4, 0);
#endif
}
#endif

// ============ WMMA GEMM: C[M,N] = A[M,K] * B[K,N]; A f16 [M][K], BT f16 [N][K], C fp32 ======
__global__ __launch_bounds__(256) void gemm_wmma_kernel(const _Float16* __restrict__ A,
                                                        const _Float16* __restrict__ BT,
                                                        float* __restrict__ C,
                                                        int M, int N, int K) {
  constexpr int BM = 128, BN = 128, BK = 32;
  __shared__ alignas(16) _Float16 As[2][BM][BK];   // 2 x 8 KB
  __shared__ alignas(16) _Float16 Bs[2][BN][BK];   // 2 x 8 KB
  int tid = threadIdx.x;
  int wave = tid >> 5, lane = tid & 31;
  int lm = lane & 15, lhi = lane >> 4;
  int bm = blockIdx.y * BM, bn = blockIdx.x * BN;
  int wm = (wave >> 1) * 32;   // 2 M-tiles per wave
  int wn = (wave & 1) * 64;    // 4 N-tiles per wave
  int nk = K / BK;
  v8f acc[2][4] = {};

#if HAS_TDM
  if (tid == 0) {
    tdm_load_2d_f16((unsigned)(unsigned long long)&As[0][0][0], A + (long)bm * K, BK, BM, K, M, K);
    tdm_load_2d_f16((unsigned)(unsigned long long)&Bs[0][0][0], BT + (long)bn * K, BK, BN, K, N, K);
  }
#endif
  for (int ks = 0; ks < nk; ++ks) {
    int buf = ks & 1;
    int kb = ks * BK;
#if HAS_TDM
    if (tid == 0) {
      if (ks + 1 < nk) {   // prefetch next stage via TDM, then wait for current stage
        tdm_load_2d_f16((unsigned)(unsigned long long)&As[buf ^ 1][0][0],
                        A + (long)bm * K + kb + BK, BK, BM, K, M, K);
        tdm_load_2d_f16((unsigned)(unsigned long long)&Bs[buf ^ 1][0][0],
                        BT + (long)bn * K + kb + BK, BK, BN, K, N, K);
        __builtin_amdgcn_s_wait_tensorcnt(2);
      } else {
        __builtin_amdgcn_s_wait_tensorcnt(0);
      }
    }
#else
    for (int e = tid; e < (BM * BK) / 8; e += 256) {
      int r = e >> 2, c8 = (e & 3) * 8;
      *(v8h*)&As[buf][r][c8] = *(const v8h*)&A[(long)(bm + r) * K + kb + c8];
    }
    for (int e = tid; e < (BN * BK) / 8; e += 256) {
      int r = e >> 2, c8 = (e & 3) * 8;
      *(v8h*)&Bs[buf][r][c8] = *(const v8h*)&BT[(long)(bn + r) * K + kb + c8];
    }
#endif
    __syncthreads();
    v16h a0 = ld_afrag(&As[buf][0][0], BK, wm, 0, lane);
    v16h a1 = ld_afrag(&As[buf][0][0], BK, wm + 16, 0, lane);
#pragma unroll
    for (int nt = 0; nt < 4; ++nt) {
      v16h bf = ld_bfrag(&Bs[buf][0][0], BK, wn + nt * 16, 0, lane);
      acc[0][nt] = wmma_f16(a0, bf, acc[0][nt]);
      acc[1][nt] = wmma_f16(a1, bf, acc[1][nt]);
    }
    __syncthreads();
  }
#pragma unroll
  for (int mt = 0; mt < 2; ++mt)
#pragma unroll
    for (int nt = 0; nt < 4; ++nt) {
      int cn = bn + wn + nt * 16 + lm;
#pragma unroll
      for (int r = 0; r < 8; ++r) {
        int cm = bm + wm + mt * 16 + r + 8 * lhi;
        C[(long)cm * N + cn] = acc[mt][nt][r];
      }
    }
}

// ============ one-time f32 -> f16 conversions ============
__global__ __launch_bounds__(256) void cvt16_kernel(const float* __restrict__ x,
                                                    _Float16* __restrict__ y) {
  long i4 = ((long)blockIdx.x * 256 + threadIdx.x) * 4;
  const float4 f = *(const float4*)&x[i4];
  v4h hv = { (_Float16)f.x, (_Float16)f.y, (_Float16)f.z, (_Float16)f.w };
  *(v4h*)&y[i4] = hv;
}

// W[K][N] fp32 -> WT[N][K] f16 (tiled transpose through LDS)
__global__ __launch_bounds__(256) void transpose16_kernel(const float* __restrict__ W,
                                                          _Float16* __restrict__ WT,
                                                          int K, int N) {
  __shared__ float tile[32][33];
  int kb = blockIdx.y * 32, nb = blockIdx.x * 32;
  int tx = threadIdx.x & 31, ty = threadIdx.x >> 5;
  for (int i = ty; i < 32; i += 8) tile[i][tx] = W[(long)(kb + i) * N + nb + tx];
  __syncthreads();
  for (int i = ty; i < 32; i += 8) WT[(long)(nb + i) * K + kb + tx] = (_Float16)tile[tx][i];
}

// ============ low-rank gate path ============
__global__ __launch_bounds__(256) void lowrank1_kernel(const float* __restrict__ h,
                                                       const float* __restrict__ W1,
                                                       float* __restrict__ tmp) {
  int idx = blockIdx.x * 256 + threadIdx.x;   // 8192*16
  int t = idx >> 4, r = idx & 15;
  const float* hp = h + (long)t * 1024;
  float s = 0.f;
  for (int i = 0; i < 1024; ++i) s += hp[i] * W1[i * 16 + r];
  tmp[idx] = s;
}

__global__ __launch_bounds__(256) void lowrank2_kernel(const float* __restrict__ tmp,
                                                       const float* __restrict__ W2,
                                                       const float* __restrict__ bias,
                                                       float* __restrict__ gk) {
  int idx = blockIdx.x * 256 + threadIdx.x;   // 8192*1024
  int t = idx >> 10, n = idx & 1023;
  const float* tp = tmp + t * 16;
  float s = bias[n];
#pragma unroll
  for (int r = 0; r < 16; ++r) s += tp[r] * W2[r * 1024 + n];
  float ls = fminf(s, 0.f) - log1pf(__expf(-fabsf(s)));   // log_sigmoid
  gk[idx] = ls * (1.0f / 16.0f);
}

// ============ gate prep: once per (b,h,chunk), fold exp gates into f16 operands ============
__global__ __launch_bounds__(256) void gate_prep_kernel(const float* __restrict__ q,
                                                        const float* __restrict__ k,
                                                        const float* __restrict__ gk,
                                                        _Float16* __restrict__ qgw,
                                                        _Float16* __restrict__ kiw,
                                                        _Float16* __restrict__ ksTw,
                                                        float* __restrict__ decw) {
  int blk = blockIdx.x;            // ((b*4+h)*32 + c)
  int c = blk & 31, bh = blk >> 5;
  int hh = bh & 3, bb = bh >> 2;
  int d = threadIdx.x;             // 0..255
  const float scale = 0.0625f;     // 256^-0.5
  long tok0 = (long)bb * 2048 + c * 64;
  long base = tok0 * 1024 + hh * 256 + d;
  float run = 0.f;
  for (int t = 0; t < 64; ++t) run += gk[base + (long)t * 1024];
  float blast = run;
  decw[(long)blk * 256 + d] = __expf(blast);
  long obase = (long)blk * (64 * 256);
  run = 0.f;
  for (int t = 0; t < 64; ++t) {
    run += gk[base + (long)t * 1024];
    float qv = q[base + (long)t * 1024];
    float kv = k[base + (long)t * 1024];
    qgw[obase + t * 256 + d] = (_Float16)(qv * __expf(run) * scale);
    kiw[obase + t * 256 + d] = (_Float16)(kv * __expf(-run));
    ksTw[obase + d * 64 + t] = (_Float16)(kv * __expf(blast - run));
  }
}

// ============ chunked GLA recurrence ============
__global__ __launch_bounds__(256) void gla_kernel(const _Float16* __restrict__ qgw,
                                                  const _Float16* __restrict__ kiw,
                                                  const _Float16* __restrict__ ksTw,
                                                  const float* __restrict__ decw,
                                                  const float* __restrict__ v,
                                                  float* __restrict__ o) {
  constexpr int HDKc = 256, CT = 64, VT = 32;
  __shared__ alignas(16) float    S[HDKc][VT];      // 32 KB fp32 state
  __shared__ alignas(16) _Float16 S16T[VT][HDKc];   // 16 KB f16 transposed shadow
  __shared__ alignas(16) _Float16 qg[CT][HDKc];     // 32 KB
  __shared__ alignas(16) _Float16 ki[CT][HDKc];     // 32 KB
  __shared__ alignas(16) _Float16 ksT[HDKc][CT];    // 32 KB
  __shared__ alignas(16) _Float16 vsT[VT][CT];      // 4 KB
  __shared__ alignas(16) _Float16 sc[CT][CT];       // 8 KB
  __shared__ alignas(16) float    decay[HDKc];      // 1 KB

  int tid = threadIdx.x;
  int wave = tid >> 5, lane = tid & 31;
  int lm = lane & 15, lhi = lane >> 4;
  int vt = blockIdx.x, hh = blockIdx.y, bb = blockIdx.z;
  int bh = bb * 4 + hh;

  for (int e = tid; e < HDKc * VT; e += 256) {
    S[e / VT][e % VT] = 0.f;
    S16T[e % VT][e / VT] = (_Float16)0.f;
  }
  __syncthreads();

  for (int c = 0; c < 32; ++c) {
    long tok0 = (long)bb * 2048 + c * 64;
    long obase = (long)(bh * 32 + c) * (64 * 256);

#if HAS_TDM
    if (tid == 0) {   // three flat 32 KB operand tiles via tensor DMA
      tdm_load_2d_f16((unsigned)(unsigned long long)&qg[0][0],  qgw + obase, 16384, 1, 16384, 1, 16384);
      tdm_load_2d_f16((unsigned)(unsigned long long)&ki[0][0],  kiw + obase, 16384, 1, 16384, 1, 16384);
      tdm_load_2d_f16((unsigned)(unsigned long long)&ksT[0][0], ksTw + obase, 16384, 1, 16384, 1, 16384);
    }
#else
    for (int e = tid; e < 2048; e += 256) {
      int off = e * 8;
      *(v8h*)(&qg[0][0] + off)  = *(const v8h*)(qgw + obase + off);
      *(v8h*)(&ki[0][0] + off)  = *(const v8h*)(kiw + obase + off);
      *(v8h*)(&ksT[0][0] + off) = *(const v8h*)(ksTw + obase + off);
    }
#endif
    decay[tid] = decw[(long)(bh * 32 + c) * 256 + tid];
    for (int e = tid; e < (CT * VT) / 4; e += 256) {   // V tile transposed [j][t]
      int t = e >> 3, j4 = (e & 7) * 4;
      const float4 f = *(const float4*)&v[(tok0 + t) * 2048 + hh * 512 + vt * VT + j4];
      vsT[j4 + 0][t] = (_Float16)f.x;
      vsT[j4 + 1][t] = (_Float16)f.y;
      vsT[j4 + 2][t] = (_Float16)f.z;
      vsT[j4 + 3][t] = (_Float16)f.w;
    }
#if HAS_TDM
    if (tid == 0) __builtin_amdgcn_s_wait_tensorcnt(0);
#endif
    __syncthreads();

    // scores = qg @ ki^T (64x64, K=256), masked tril
#pragma unroll
    for (int s = 0; s < 2; ++s) {
      int tt = wave * 2 + s;
      int tm = tt >> 2, tn = tt & 3;
      v8f acc = {};
#pragma unroll
      for (int kk = 0; kk < 8; ++kk) {
        v16h a  = ld_afrag(&qg[0][0], HDKc, tm * 16, kk * 32, lane);
        v16h bf = ld_bfrag(&ki[0][0], HDKc, tn * 16, kk * 32, lane);
        acc = wmma_f16(a, bf, acc);
      }
      int n = tn * 16 + lm;
#pragma unroll
      for (int r = 0; r < 8; ++r) {
        int m = tm * 16 + r + 8 * lhi;
        sc[m][n] = (_Float16)((n <= m) ? acc[r] : 0.f);
      }
    }
    __syncthreads();

    // o = sc @ v (K=64) + qg @ S (K=256)
    {
      int tm = wave >> 1, tn = wave & 1;
      v8f acc = {};
#pragma unroll
      for (int kk = 0; kk < 2; ++kk) {
        v16h a  = ld_afrag(&sc[0][0], CT, tm * 16, kk * 32, lane);
        v16h bf = ld_bfrag(&vsT[0][0], CT, tn * 16, kk * 32, lane);
        acc = wmma_f16(a, bf, acc);
      }
#pragma unroll
      for (int kk = 0; kk < 8; ++kk) {
        v16h a  = ld_afrag(&qg[0][0], HDKc, tm * 16, kk * 32, lane);
        v16h bf = ld_bfrag(&S16T[0][0], HDKc, tn * 16, kk * 32, lane);
        acc = wmma_f16(a, bf, acc);
      }
      int n = tn * 16 + lm;
#pragma unroll
      for (int r = 0; r < 8; ++r) {
        int m = tm * 16 + r + 8 * lhi;
        o[(tok0 + m) * 2048 + hh * 512 + vt * VT + n] = acc[r];
      }
    }
    __syncthreads();

    // S = S*decay + ks^T @ v; refresh f16 shadow
#pragma unroll
    for (int s = 0; s < 4; ++s) {
      int tt = wave * 4 + s;
      int td = tt >> 1, tj = tt & 1;
      v8f acc = {};
#pragma unroll
      for (int kk = 0; kk < 2; ++kk) {
        v16h a  = ld_afrag(&ksT[0][0], CT, td * 16, kk * 32, lane);
        v16h bf = ld_bfrag(&vsT[0][0], CT, tj * 16, kk * 32, lane);
        acc = wmma_f16(a, bf, acc);
      }
      int n = tj * 16 + lm;
#pragma unroll
      for (int r = 0; r < 8; ++r) {
        int d = td * 16 + r + 8 * lhi;
        float sn = S[d][n] * decay[d] + acc[r];
        S[d][n] = sn;
        S16T[n][d] = (_Float16)sn;
      }
    }
    __syncthreads();
  }
}

// ============ RMS-norm * g_norm_w * SiLU(g); emits f16 activation for Wo GEMM ============
__global__ __launch_bounds__(256) void gate_norm_kernel(const float* __restrict__ o,
                                                        const float* __restrict__ g,
                                                        const float* __restrict__ gw,
                                                        _Float16* __restrict__ o16) {
  __shared__ float red[256];
  int tid = threadIdx.x;
  long base = (long)(blockIdx.x >> 2) * 2048 + (blockIdx.x & 3) * 512;
  float x0 = o[base + tid], x1 = o[base + 256 + tid];
  red[tid] = x0 * x0 + x1 * x1;
  __syncthreads();
  for (int s = 128; s > 0; s >>= 1) {
    if (tid < s) red[tid] += red[tid + s];
    __syncthreads();
  }
  float rinv = rsqrtf(red[0] * (1.0f / 512.0f) + 1e-5f);
#pragma unroll
  for (int ii = 0; ii < 2; ++ii) {
    int i = tid + ii * 256;
    float gv = g[base + i];
    float sig = 1.f / (1.f + __expf(-gv));
    o16[base + i] = (_Float16)(o[base + i] * rinv * gw[i] * gv * sig);
  }
}

extern "C" void kernel_launch(void* const* d_in, const int* in_sizes, int n_in,
                              void* d_out, int out_size, void* d_ws, size_t ws_size,
                              hipStream_t stream) {
  const float* h    = (const float*)d_in[0];
  const float* Wq   = (const float*)d_in[1];
  const float* Wk   = (const float*)d_in[2];
  const float* Wv   = (const float*)d_in[3];
  const float* Wg   = (const float*)d_in[4];
  const float* Wgk1 = (const float*)d_in[5];
  const float* Wgk2 = (const float*)d_in[6];
  const float* bgk2 = (const float*)d_in[7];
  const float* gnw  = (const float*)d_in[8];
  const float* Wo   = (const float*)d_in[9];
  float* out = (float*)d_out;

  const long T = 8192;
  float* ws   = (float*)d_ws;
  float* qb   = ws;                  // T*1024 f32
  float* kb   = qb + T * 1024;
  float* gkb  = kb + T * 1024;
  float* vb   = gkb + T * 1024;      // T*2048 f32
  float* gb   = vb + T * 2048;
  float* ob   = gb + T * 2048;
  float* tmpb = ob + T * 2048;       // T*16
  float* decw = tmpb + T * 16;       // 512*256
  _Float16* qgw   = (_Float16*)(decw + 512 * 256);   // T*1024 f16 each
  _Float16* kiw   = qgw + T * 1024;
  _Float16* ksTw  = kiw + T * 1024;
  _Float16* h16   = ksTw + T * 1024;                 // T*1024
  _Float16* ob16  = h16 + T * 1024;                  // T*2048
  _Float16* WqT16 = ob16 + T * 2048;                 // 1024*1024
  _Float16* WkT16 = WqT16 + 1024 * 1024;             // 1024*1024
  _Float16* WvT16 = WkT16 + 1024 * 1024;             // 2048*1024
  _Float16* WgT16 = WvT16 + 2048 * 1024;             // 2048*1024
  _Float16* WoT16 = WgT16 + 2048 * 1024;             // 1024*2048

  // one-time conversions: activations to f16, weights to transposed f16
  cvt16_kernel<<<(T * 1024) / 1024, 256, 0, stream>>>(h, h16);
  transpose16_kernel<<<dim3(1024 / 32, 1024 / 32), 256, 0, stream>>>(Wq, WqT16, 1024, 1024);
  transpose16_kernel<<<dim3(1024 / 32, 1024 / 32), 256, 0, stream>>>(Wk, WkT16, 1024, 1024);
  transpose16_kernel<<<dim3(2048 / 32, 1024 / 32), 256, 0, stream>>>(Wv, WvT16, 1024, 2048);
  transpose16_kernel<<<dim3(2048 / 32, 1024 / 32), 256, 0, stream>>>(Wg, WgT16, 1024, 2048);
  transpose16_kernel<<<dim3(1024 / 32, 2048 / 32), 256, 0, stream>>>(Wo, WoT16, 2048, 1024);

  // projections (TDM-staged WMMA GEMMs)
  gemm_wmma_kernel<<<dim3(1024 / 128, 8192 / 128), 256, 0, stream>>>(h16, WqT16, qb, 8192, 1024, 1024);
  gemm_wmma_kernel<<<dim3(1024 / 128, 8192 / 128), 256, 0, stream>>>(h16, WkT16, kb, 8192, 1024, 1024);
  gemm_wmma_kernel<<<dim3(2048 / 128, 8192 / 128), 256, 0, stream>>>(h16, WvT16, vb, 8192, 2048, 1024);
  gemm_wmma_kernel<<<dim3(2048 / 128, 8192 / 128), 256, 0, stream>>>(h16, WgT16, gb, 8192, 2048, 1024);

  // low-rank gate: gk = log_sigmoid(h@Wgk1@Wgk2 + b)/16
  lowrank1_kernel<<<(8192 * 16) / 256, 256, 0, stream>>>(h, Wgk1, tmpb);
  lowrank2_kernel<<<(8192 * 1024) / 256, 256, 0, stream>>>(tmpb, Wgk2, bgk2, gkb);

  // fold gates into f16 operands once per (b,h,chunk)
  gate_prep_kernel<<<512, 256, 0, stream>>>(qb, kb, gkb, qgw, kiw, ksTw, decw);

  // chunked gated linear attention
  gla_kernel<<<dim3(16, 4, 4), 256, 0, stream>>>(qgw, kiw, ksTw, decw, vb, ob);

  // RMSNorm * g_norm_w * SiLU(g) -> f16 activation
  gate_norm_kernel<<<8192 * 4, 256, 0, stream>>>(ob, gb, gnw, ob16);

  // output projection
  gemm_wmma_kernel<<<dim3(1024 / 128, 8192 / 128), 256, 0, stream>>>(ob16, WoT16, out, 8192, 1024, 2048);
}
